// GRUStack_37735582662719
// MI455X (gfx1250) — matrix operand order repaired and checked by reading, not compiled, not run on hardware
//
#include <hip/hip_runtime.h>
#include <hip/hip_bf16.h>
#include <math.h>

// ---------------------------------------------------------------------------
// GRU stack for MI455X (gfx1250, wave32, WMMA).
// - GEMMs: v_wmma_f32_16x16x32_bf16 (fp32 accumulate), weight tiles staged to
//   LDS with double-buffered GLOBAL_LOAD_ASYNC_TO_LDS_B128 (ASYNCcnt).
// - Recurrent scan: persistent 24-block kernel, bf16 W_hh slice async-copied
//   into LDS once (64KB of the 320KB WGP pool), grid-synced twice per step.
// ---------------------------------------------------------------------------

typedef __attribute__((ext_vector_type(16))) __bf16       bf16x16;
typedef __attribute__((ext_vector_type(8)))  float        f32x8;
typedef __attribute__((ext_vector_type(4)))  unsigned int u32x4;

#define D_IN     768
#define D_MODEL  512
#define GATES    1536
#define N_LAYERS 4
#define D_OUT    768
#define BATCH    64
#define SEQ      1024
#define ROWS     (BATCH * SEQ)      // 65536
#define NBLK_SCAN 24                // 1536 / 64 gate columns per block
#define KCHUNK   128                // K-chunk staged to LDS per double-buffer slot

#define ASYNC_WAIT(N) asm volatile("s_wait_asynccnt " #N ::: "memory")

// Async global->LDS copy of 16 bytes per lane. the LDS address is taken from loff;
// low 32 bits of a generic pointer to __shared__ are the workgroup LDS offset.
__device__ __forceinline__ void async_b128(unsigned short* lds_dst,
                                           const unsigned short* gsrc) {
  unsigned loff = (unsigned)(unsigned long long)(uintptr_t)lds_dst;
  asm volatile("global_load_async_to_lds_b128 %0, %1, off"
               :: "v"(loff), "v"(gsrc) : "memory");
}

__device__ __forceinline__ unsigned short f32_to_bf16_rn(float f) {
  union { float f; unsigned int u; } v; v.f = f;
  unsigned int lsb = (v.u >> 16) & 1u;
  v.u += 0x7fffu + lsb;             // round-to-nearest-even
  return (unsigned short)(v.u >> 16);
}

// A fragment: 16x32 (MxK) bf16, row-major source with leading dim ld.
// Lane l<16: row l, K = {0..7, 16..23}; lane l>=16: row l-16, K = {8..15, 24..31}.
__device__ __forceinline__ bf16x16 load_frag_a(const unsigned short* __restrict__ A,
                                               int ld, int lane) {
  const int row = lane & 15;
  const int kh  = (lane >> 4) << 3;          // 0 or 8
  const unsigned short* r = A + row * ld;
  union { bf16x16 v; u32x4 q[2]; } u;
  u.q[0] = *(const u32x4*)(r + kh);          // K = kh .. kh+7
  u.q[1] = *(const u32x4*)(r + 16 + kh);     // K = 16+kh .. 16+kh+7
  return u.v;
}

// B fragment: 32x16 (KxN) bf16. Column n of B = row n of the (NxK) weight.
// Lane n<16: column n, K = {0..7, 16..23}; lane n>=16: column n-16, K = {8..15, 24..31}.
__device__ __forceinline__ bf16x16 load_frag_b(const unsigned short* __restrict__ W,
                                               int ld, int lane) {
  const int col = lane & 15;
  const int kh  = (lane >> 4) << 3;          // 0 or 8
  const unsigned short* r = W + col * ld + kh;
  union { bf16x16 v; u32x4 q[2]; } u;
  u.q[0] = *(const u32x4*)(r);
  u.q[1] = *(const u32x4*)(r + 16);
  return u.v;
}

__device__ __forceinline__ f32x8 wmma_bf16(bf16x16 a, bf16x16 b, f32x8 c) {
  return __builtin_amdgcn_wmma_f32_16x16x32_bf16(
      /*neg_a=*/false, a, /*neg_b=*/false, b,
      /*c_mod=*/(short)0, c, /*reuse_a=*/false, /*reuse_b=*/false);
}

// ------------------------- fp32 -> bf16 convert ----------------------------
__global__ void __launch_bounds__(256)
cvt_f32_bf16(const float* __restrict__ in, unsigned short* __restrict__ out, size_t n) {
  size_t i = (size_t)blockIdx.x * blockDim.x + threadIdx.x;
  const size_t stride = (size_t)gridDim.x * blockDim.x;
  for (; i < n; i += stride) out[i] = f32_to_bf16_rn(in[i]);
}

// --------------------- generic bf16 WMMA GEMM ------------------------------
// C[M x N] = A[M x K] (bf16, row-major) @ W[N x K]^T (bf16, row-major) + bias
// Block: 256 thr (8 waves), each wave a 16x64 tile; block tile 128x64.
// B tile (shared by all 8 waves) staged to LDS via double-buffered async
// copies in K-chunks of 128. Grid: (N/64, ceil(M/128)).  K % 128 == 0.
__global__ void __launch_bounds__(256)
gemm_bf16(const unsigned short* __restrict__ A, const unsigned short* __restrict__ W,
          const float* __restrict__ bias, float* __restrict__ C,
          int M, int N, int K) {
  __shared__ unsigned short sB[2][64 * KCHUNK];   // 2 x 16 KB
  const int wave = threadIdx.x >> 5;
  const int lane = threadIdx.x & 31;
  const int m0 = blockIdx.y * 128 + wave * 16;
  const int n0 = blockIdx.x * 64;
  const bool active = (m0 < M);                   // wave-uniform
  const int nchunks = K / KCHUNK;

  // stage chunk 0: 64 cols x 128 k x bf16 = 16KB = 1024 x b128; 4 per thread
#pragma unroll
  for (int j = 0; j < 4; ++j) {
    const int e   = (threadIdx.x + 256 * j) * 8;
    const int col = e >> 7;
    const int kk  = e & 127;
    async_b128(&sB[0][col * KCHUNK + kk], W + (size_t)(n0 + col) * K + kk);
  }

  f32x8 acc0 = {}, acc1 = {}, acc2 = {}, acc3 = {};
  const unsigned short* Ar = A + (size_t)m0 * K;

  for (int c = 0; c < nchunks; ++c) {
    if (c + 1 < nchunks) {
      const int k0 = (c + 1) * KCHUNK;
#pragma unroll
      for (int j = 0; j < 4; ++j) {
        const int e   = (threadIdx.x + 256 * j) * 8;
        const int col = e >> 7;
        const int kk  = e & 127;
        async_b128(&sB[(c + 1) & 1][col * KCHUNK + kk],
                   W + (size_t)(n0 + col) * K + k0 + kk);
      }
      ASYNC_WAIT(4);     // in-order: chunk c's 4 copies done, chunk c+1 in flight
    } else {
      ASYNC_WAIT(0);
    }
    __syncthreads();
    if (active) {
      const unsigned short* sb = sB[c & 1];
#pragma unroll
      for (int k = 0; k < KCHUNK; k += 32) {
        bf16x16 a  = load_frag_a(Ar + c * KCHUNK + k, K, lane);
        bf16x16 b0 = load_frag_b(sb +  0 * KCHUNK + k, KCHUNK, lane);
        bf16x16 b1 = load_frag_b(sb + 16 * KCHUNK + k, KCHUNK, lane);
        bf16x16 b2 = load_frag_b(sb + 32 * KCHUNK + k, KCHUNK, lane);
        bf16x16 b3 = load_frag_b(sb + 48 * KCHUNK + k, KCHUNK, lane);
        acc0 = wmma_bf16(a, b0, acc0);
        acc1 = wmma_bf16(a, b1, acc1);
        acc2 = wmma_bf16(a, b2, acc2);
        acc3 = wmma_bf16(a, b3, acc3);
      }
    }
    __syncthreads();     // buffer (c&1) free for reuse at iteration c+2
  }

  if (active) {
    const int n  = lane & 15;
    const int mA = (lane >> 4) << 3;           // +8 rows for upper half-wave
    const float bv0 = bias ? bias[n0 +  0 + n] : 0.f;
    const float bv1 = bias ? bias[n0 + 16 + n] : 0.f;
    const float bv2 = bias ? bias[n0 + 32 + n] : 0.f;
    const float bv3 = bias ? bias[n0 + 48 + n] : 0.f;
#pragma unroll
    for (int v = 0; v < 8; ++v) {
      float* Cr = C + (size_t)(m0 + mA + v) * N + n0 + n;
      Cr[ 0] = acc0[v] + bv0;
      Cr[16] = acc1[v] + bv1;
      Cr[32] = acc2[v] + bv2;
      Cr[48] = acc3[v] + bv3;
    }
  }
}

// ------------------------- grid-wide barrier -------------------------------
__device__ __forceinline__ void grid_barrier(int* c, int nblk) {
  __syncthreads();
  if (threadIdx.x == 0) {
    __threadfence();
    __hip_atomic_fetch_add(c, 1, __ATOMIC_ACQ_REL, __HIP_MEMORY_SCOPE_AGENT);
    while (__hip_atomic_load(c, __ATOMIC_ACQUIRE, __HIP_MEMORY_SCOPE_AGENT) < nblk)
      __builtin_amdgcn_s_sleep(1);
  }
  __syncthreads();
}

// ----------------------- persistent GRU scan -------------------------------
// 24 blocks x 128 thr (4 waves). Block b owns gate columns [64b, 64b+64).
// Its bf16 W_hh slice (64x512 = 64KB) is async-copied into LDS once and
// reused for all 1024 timesteps. h_{t-1} lives in a 64KB global bf16 buffer.
__global__ void __launch_bounds__(128)
gru_scan(const unsigned short* __restrict__ Whh_bf,   // [1536][512] bf16 (layer)
         const float* __restrict__ bhh,               // [1536]
         const float* __restrict__ xg,                // [ROWS][1536] fp32
         float* __restrict__ hgbuf,                   // [64][1536]
         float* __restrict__ hprev_f,                 // [64][512]  (zeroed)
         unsigned short* __restrict__ hprev_bf,       // [64][512]  (zeroed)
         float* __restrict__ ys,                      // [ROWS][512]
         int* __restrict__ counters)                  // [2*SEQ]    (zeroed)
{
  __shared__ unsigned short sW[64 * D_MODEL];      // 64 KB of 320 KB WGP LDS
  const int tid  = threadIdx.x;
  const int wave = tid >> 5;
  const int lane = tid & 31;
  const int gc0  = blockIdx.x * 64;                // gate-column base

  // async-stage this block's W_hh slice: 4096 x b128, 32 per thread
#pragma unroll 4
  for (int j = 0; j < 32; ++j) {
    const int e   = (tid + 128 * j) * 8;
    const int col = e >> 9;
    const int kk  = e & 511;
    async_b128(&sW[col * D_MODEL + kk],
               Whh_bf + (size_t)(gc0 + col) * D_MODEL + kk);
  }
  ASYNC_WAIT(0);
  __syncthreads();

  const int m0 = wave * 16;                        // batch rows 0..63, 16/wave
  const int n  = lane & 15;
  const int mA = (lane >> 4) << 3;
  const float bb0 = bhh[gc0 +  0 + n];
  const float bb1 = bhh[gc0 + 16 + n];
  const float bb2 = bhh[gc0 + 32 + n];
  const float bb3 = bhh[gc0 + 48 + n];
  const int gtid = blockIdx.x * 128 + tid;
  const int nthr = NBLK_SCAN * 128;

  for (int t = 0; t < SEQ; ++t) {
    // ---- stage 1: hg[64 x 64-col tile] = h_{t-1} @ Whh_slice^T + bhh ----
    f32x8 acc0 = {}, acc1 = {}, acc2 = {}, acc3 = {};
    for (int k = 0; k < D_MODEL; k += 32) {
      bf16x16 a  = load_frag_a(hprev_bf + m0 * D_MODEL + k, D_MODEL, lane);
      bf16x16 b0 = load_frag_b(sW +  0 * D_MODEL + k, D_MODEL, lane);
      bf16x16 b1 = load_frag_b(sW + 16 * D_MODEL + k, D_MODEL, lane);
      bf16x16 b2 = load_frag_b(sW + 32 * D_MODEL + k, D_MODEL, lane);
      bf16x16 b3 = load_frag_b(sW + 48 * D_MODEL + k, D_MODEL, lane);
      acc0 = wmma_bf16(a, b0, acc0);
      acc1 = wmma_bf16(a, b1, acc1);
      acc2 = wmma_bf16(a, b2, acc2);
      acc3 = wmma_bf16(a, b3, acc3);
    }
#pragma unroll
    for (int v = 0; v < 8; ++v) {
      float* o = hgbuf + (size_t)(m0 + mA + v) * GATES + gc0 + n;
      o[ 0] = acc0[v] + bb0;
      o[16] = acc1[v] + bb1;
      o[32] = acc2[v] + bb2;
      o[48] = acc3[v] + bb3;
    }
    grid_barrier(&counters[2 * t], NBLK_SCAN);

    // ---- stage 2: gates + state update (all blocks cooperate) ----
    for (int i = gtid; i < BATCH * D_MODEL; i += nthr) {
      const int b = i >> 9, j = i & 511;
      const float* xr = xg    + (size_t)(b * SEQ + t) * GATES;
      const float* hr = hgbuf + (size_t)b * GATES;
      float r  = 1.f / (1.f + expf(-(xr[j]       + hr[j])));
      float z  = 1.f / (1.f + expf(-(xr[512 + j] + hr[512 + j])));
      float nn = tanhf(xr[1024 + j] + r * hr[1024 + j]);
      float hn = (1.f - z) * nn + z * hprev_f[i];
      ys[(size_t)(b * SEQ + t) * D_MODEL + j] = hn;
      hprev_f[i]  = hn;
      hprev_bf[i] = f32_to_bf16_rn(hn);
    }
    grid_barrier(&counters[2 * t + 1], NBLK_SCAN);
  }
}

// ------------------- residual add + LayerNorm (in place) -------------------
__global__ void __launch_bounds__(256)
add_ln(float* __restrict__ h, const float* __restrict__ go,
       const float* __restrict__ g, const float* __restrict__ b, int rows) {
  const int row  = blockIdx.x * 8 + (threadIdx.x >> 5);
  const int lane = threadIdx.x & 31;
  if (row >= rows) return;
  float* hr = h + (size_t)row * D_MODEL;
  const float* gr = go + (size_t)row * D_MODEL;
  float v[16]; float s = 0.f;
#pragma unroll
  for (int i = 0; i < 16; ++i) { v[i] = hr[lane + 32 * i] + gr[lane + 32 * i]; s += v[i]; }
#pragma unroll
  for (int off = 16; off; off >>= 1) s += __shfl_xor(s, off, 32);
  const float mu = s * (1.f / 512.f);
  float q = 0.f;
#pragma unroll
  for (int i = 0; i < 16; ++i) { float d = v[i] - mu; q += d * d; }
#pragma unroll
  for (int off = 16; off; off >>= 1) q += __shfl_xor(q, off, 32);
  const float rs = rsqrtf(q * (1.f / 512.f) + 1e-5f);
#pragma unroll
  for (int i = 0; i < 16; ++i)
    hr[lane + 32 * i] = (v[i] - mu) * rs * g[lane + 32 * i] + b[lane + 32 * i];
}

// ---------------------- gather last timestep (bf16) ------------------------
__global__ void __launch_bounds__(256)
gather_last(const float* __restrict__ h, unsigned short* __restrict__ last_bf) {
  const int i = blockIdx.x * 256 + threadIdx.x;
  if (i < BATCH * D_MODEL) {
    const int b = i >> 9, j = i & 511;
    last_bf[i] = f32_to_bf16_rn(h[(size_t)(b * SEQ + SEQ - 1) * D_MODEL + j]);
  }
}

// --------------------------- cosine normalize ------------------------------
__global__ void __launch_bounds__(256)
cosnorm(const float* __restrict__ raw, float* __restrict__ out) {
  const int row  = blockIdx.x * 8 + (threadIdx.x >> 5);
  const int lane = threadIdx.x & 31;
  if (row >= BATCH) return;
  float v[24]; float s = 0.f;
#pragma unroll
  for (int i = 0; i < 24; ++i) { v[i] = raw[(size_t)row * D_OUT + lane + 32 * i]; s += v[i] * v[i]; }
#pragma unroll
  for (int off = 16; off; off >>= 1) s += __shfl_xor(s, off, 32);
  const float inv = 1.f / fmaxf(sqrtf(s), 1e-12f);
#pragma unroll
  for (int i = 0; i < 24; ++i) out[(size_t)row * D_OUT + lane + 32 * i] = v[i] * inv;
}

// ---------------------------------------------------------------------------
extern "C" void kernel_launch(void* const* d_in, const int* in_sizes, int n_in,
                              void* d_out, int out_size, void* d_ws, size_t ws_size,
                              hipStream_t stream) {
  const float* x    = (const float*)d_in[0];
  const float* inW  = (const float*)d_in[1];
  const float* inb  = (const float*)d_in[2];
  const float* gWih = (const float*)d_in[3];
  const float* gWhh = (const float*)d_in[4];
  const float* gbih = (const float*)d_in[5];
  const float* gbhh = (const float*)d_in[6];
  const float* lng  = (const float*)d_in[7];
  const float* lnb  = (const float*)d_in[8];
  const float* outW = (const float*)d_in[9];
  const float* outb = (const float*)d_in[10];
  float* out = (float*)d_out;

  char* ws = (char*)d_ws;
  size_t off = 0;
  auto alloc = [&](size_t bytes) -> void* {
    void* p = ws + off;
    off += (bytes + 255) & ~(size_t)255;
    return p;
  };
  unsigned short* x_bf    = (unsigned short*)alloc((size_t)ROWS * D_IN * 2);
  unsigned short* inW_bf  = (unsigned short*)alloc((size_t)D_MODEL * D_IN * 2);
  unsigned short* Wih_bf  = (unsigned short*)alloc((size_t)N_LAYERS * GATES * D_MODEL * 2);
  unsigned short* Whh_bf  = (unsigned short*)alloc((size_t)N_LAYERS * GATES * D_MODEL * 2);
  unsigned short* outW_bf = (unsigned short*)alloc((size_t)D_OUT * D_MODEL * 2);
  float*          h       = (float*)alloc((size_t)ROWS * D_MODEL * 4);
  unsigned short* h_bf    = (unsigned short*)alloc((size_t)ROWS * D_MODEL * 2);
  float*          xg      = (float*)alloc((size_t)ROWS * GATES * 4);
  float*          go      = (float*)alloc((size_t)ROWS * D_MODEL * 4);
  float*          hgbuf   = (float*)alloc((size_t)BATCH * GATES * 4);
  float*          hprev_f = (float*)alloc((size_t)BATCH * D_MODEL * 4);
  unsigned short* hprev_bf= (unsigned short*)alloc((size_t)BATCH * D_MODEL * 2);
  unsigned short* last_bf = (unsigned short*)alloc((size_t)BATCH * D_MODEL * 2);
  float*          raw     = (float*)alloc((size_t)BATCH * D_OUT * 4);
  int*            counters= (int*)alloc((size_t)2 * SEQ * 4);

  // Stage inputs/weights as bf16 for the WMMA path.
  cvt_f32_bf16<<<8192, 256, 0, stream>>>(x, x_bf, (size_t)ROWS * D_IN);
  cvt_f32_bf16<<<1536, 256, 0, stream>>>(inW, inW_bf, (size_t)D_MODEL * D_IN);
  cvt_f32_bf16<<<4096, 256, 0, stream>>>(gWih, Wih_bf, (size_t)N_LAYERS * GATES * D_MODEL);
  cvt_f32_bf16<<<4096, 256, 0, stream>>>(gWhh, Whh_bf, (size_t)N_LAYERS * GATES * D_MODEL);
  cvt_f32_bf16<<<1536, 256, 0, stream>>>(outW, outW_bf, (size_t)D_OUT * D_MODEL);

  // h = x @ in_W^T + in_b  (M=65536, N=512, K=768)
  gemm_bf16<<<dim3(D_MODEL / 64, ROWS / 128), 256, 0, stream>>>(
      x_bf, inW_bf, inb, h, ROWS, D_MODEL, D_IN);

  for (int l = 0; l < N_LAYERS; ++l) {
    cvt_f32_bf16<<<8192, 256, 0, stream>>>(h, h_bf, (size_t)ROWS * D_MODEL);
    // xg = h @ W_ih^T + b_ih  (M=65536, N=1536, K=512)
    gemm_bf16<<<dim3(GATES / 64, ROWS / 128), 256, 0, stream>>>(
        h_bf, Wih_bf + (size_t)l * GATES * D_MODEL, gbih + (size_t)l * GATES,
        xg, ROWS, GATES, D_MODEL);
    // reset scan state (h0 = 0) and one-shot barrier counters
    hipMemsetAsync(counters, 0, (size_t)2 * SEQ * 4, stream);
    hipMemsetAsync(hprev_f, 0, (size_t)BATCH * D_MODEL * 4, stream);
    hipMemsetAsync(hprev_bf, 0, (size_t)BATCH * D_MODEL * 2, stream);
    gru_scan<<<NBLK_SCAN, 128, 0, stream>>>(
        Whh_bf + (size_t)l * GATES * D_MODEL, gbhh + (size_t)l * GATES,
        xg, hgbuf, hprev_f, hprev_bf, go, counters);
    // h = LayerNorm(h + gru_out)
    add_ln<<<ROWS / 8, 256, 0, stream>>>(h, go, lng + (size_t)l * D_MODEL,
                                         lnb + (size_t)l * D_MODEL, ROWS);
  }

  gather_last<<<(BATCH * D_MODEL) / 256, 256, 0, stream>>>(h, last_bf);
  // raw = last @ out_W^T + out_b  (M=64, N=768, K=512)
  gemm_bf16<<<dim3(D_OUT / 64, (BATCH + 127) / 128), 256, 0, stream>>>(
      last_bf, outW_bf, outb, raw, BATCH, D_OUT, D_MODEL);
  cosnorm<<<(BATCH + 7) / 8, 256, 0, stream>>>(raw, out);
}
